// Decoder_18519898980864
// MI455X (gfx1250) — matrix operand order repaired and checked by reading, not compiled, not run on hardware
//
#include <hip/hip_runtime.h>
#include <hip/hip_bf16.h>

// Problem constants (from the reference file)
static constexpr int Bsz  = 256;   // batch
static constexpr int DIN  = 512;
static constexpr int Hdim = 1024;
static constexpr int DOUT = 512;
static constexpr int Tlen = 256;
static constexpr int GN   = 6 * Hdim;   // 6144 gate columns (w_ih ++ w_hh)

typedef __attribute__((ext_vector_type(16))) __bf16 v16bf;
typedef __attribute__((ext_vector_type(8)))  float  v8f;

union Frag16 { v16bf v; uint4 q[2]; };

__device__ __forceinline__ unsigned short f2bf(float f) {
    unsigned u = __float_as_uint(f);
    // round-to-nearest-even bf16
    return (unsigned short)((u + 0x7FFFu + ((u >> 16) & 1u)) >> 16);
}

// ---------------------------------------------------------------------------
// Elementwise fp32 -> bf16 conversion (weights / activations)
// ---------------------------------------------------------------------------
__global__ void k_f32_to_bf16(const float* __restrict__ in,
                              unsigned short* __restrict__ out, int n) {
    int i = blockIdx.x * blockDim.x + threadIdx.x;
    if (i < n) out[i] = f2bf(in[i]);
}

// h0 = fcn_latent: copy to d_out latent slot and make bf16 copy for WMMA
__global__ void k_h_init(const float* __restrict__ hf,
                         unsigned short* __restrict__ hb,
                         float* __restrict__ latent_out, int n) {
    int i = blockIdx.x * blockDim.x + threadIdx.x;
    if (i < n) {
        float v = hf[i];
        latent_out[i] = v;
        hb[i] = f2bf(v);
    }
}

// ---------------------------------------------------------------------------
// C[M x ldc] = A[M x K](bf16, row-major) * W[N x K](bf16, row-major)^T (+bias)
// fp32 accumulation via v_wmma_f32_16x16x32_bf16.
// Block: 256 threads = 8 waves; block tile 64(M) x 256(N); wave tile 32x64.
// grid.x = N/256, grid.y = M/64.  K % 32 == 0, tiles divide exactly.
// ---------------------------------------------------------------------------
__global__ __launch_bounds__(256) void k_gemm_bf16_tn(
    const __bf16* __restrict__ A, const __bf16* __restrict__ W,
    float* __restrict__ C, const float* __restrict__ bias,
    int K, int ldc)
{
    const int lane = threadIdx.x & 31;
    const int wave = threadIdx.x >> 5;
    const int r    = lane & 15;   // row within 16-row fragment
    const int hi   = lane >> 4;   // K-half select (ISA 16-bit A/B layouts)

    const int mTile = blockIdx.y * 64  + (wave >> 2) * 32;
    const int nTile = blockIdx.x * 256 + (wave & 3) * 64;

    const uint4* __restrict__ Aq = (const uint4*)A;
    const uint4* __restrict__ Wq = (const uint4*)W;
    const int Kq = K >> 3;                 // uint4 (8 bf16) per row

    int aBase[2];
    #pragma unroll
    for (int i = 0; i < 2; ++i)
        aBase[i] = (mTile + 16 * i + r) * Kq + hi;      // A: chunks at k+8*hi, +16
    int wBase[4];
    #pragma unroll
    for (int j = 0; j < 4; ++j)
        wBase[j] = (nTile + 16 * j + r) * Kq + hi;      // B: chunks at k+8*hi, +16

    v8f acc[2][4] = {};

    for (int kq = 0; kq < Kq; kq += 4) {   // 32 K-elements per iteration
        Frag16 a[2], b[4];
        #pragma unroll
        for (int i = 0; i < 2; ++i) {
            a[i].q[0] = Aq[aBase[i] + kq];
            a[i].q[1] = Aq[aBase[i] + kq + 2];
        }
        #pragma unroll
        for (int j = 0; j < 4; ++j) {
            b[j].q[0] = Wq[wBase[j] + kq];
            b[j].q[1] = Wq[wBase[j] + kq + 2];
        }
        #pragma unroll
        for (int i = 0; i < 2; ++i)
            #pragma unroll
            for (int j = 0; j < 4; ++j)
                acc[i][j] = __builtin_amdgcn_wmma_f32_16x16x32_bf16(
                    false, a[i].v, false, b[j].v,
                    (short)0, acc[i][j], false, false);
    }

    #pragma unroll
    for (int i = 0; i < 2; ++i) {
        #pragma unroll
        for (int j = 0; j < 4; ++j) {
            const int col = nTile + 16 * j + r;
            const float bj = bias ? bias[col] : 0.0f;
            #pragma unroll
            for (int v = 0; v < 8; ++v) {
                const int row = mTile + 16 * i + 8 * hi + v;   // ISA C/D layout
                C[(size_t)row * ldc + col] = acc[i][j][v] + bj;
            }
        }
    }
}

// ---------------------------------------------------------------------------
// GRU gating: gates[B x 6144] = [i_r i_z i_n | h_r h_z h_n] (pre-bias).
// Updates h (fp32, in-place) and writes bf16 copy for the next step's GEMMs.
// ---------------------------------------------------------------------------
__global__ __launch_bounds__(256) void k_gru_gate(
    const float* __restrict__ gates,
    const float* __restrict__ b_ih, const float* __restrict__ b_hh,
    float* __restrict__ h, unsigned short* __restrict__ hb)
{
    int i = blockIdx.x * blockDim.x + threadIdx.x;   // 0 .. B*H-1
    int row = i >> 10;
    int col = i & (Hdim - 1);
    const float* g = gates + (size_t)row * GN;

    float ir = g[col]            + b_ih[col];
    float iz = g[col + 1024]     + b_ih[col + 1024];
    float inn= g[col + 2048]     + b_ih[col + 2048];
    float hr = g[col + 3072]     + b_hh[col];
    float hz = g[col + 4096]     + b_hh[col + 1024];
    float hn = g[col + 5120]     + b_hh[col + 2048];

    float rg = 1.0f / (1.0f + __expf(-(ir + hr)));
    float zg = 1.0f / (1.0f + __expf(-(iz + hz)));
    float ng = tanhf(inn + rg * hn);
    float hnew = (1.0f - zg) * ng + zg * h[i];

    h[i]  = hnew;
    hb[i] = f2bf(hnew);
}

// ---------------------------------------------------------------------------
// Host-side orchestration (graph-capturable: everything on `stream`)
// ---------------------------------------------------------------------------
extern "C" void kernel_launch(void* const* d_in, const int* in_sizes, int n_in,
                              void* d_out, int out_size, void* d_ws, size_t ws_size,
                              hipStream_t stream) {
    const float* x      = (const float*)d_in[0];
    const float* fcn_w1 = (const float*)d_in[1];
    const float* fcn_b1 = (const float*)d_in[2];
    const float* fcn_w2 = (const float*)d_in[3];
    const float* fcn_b2 = (const float*)d_in[4];
    const float* w_ih   = (const float*)d_in[5];
    const float* b_ih   = (const float*)d_in[6];
    const float* w_hh   = (const float*)d_in[7];
    const float* b_hh   = (const float*)d_in[8];
    const float* out_w  = (const float*)d_in[9];
    const float* out_b  = (const float*)d_in[10];

    // Workspace layout (bf16 buffers stored as ushort bits), 256B aligned.
    char* ws = (char*)d_ws;
    size_t off = 0;
    auto alloc = [&](size_t bytes) { char* p = ws + off; off = (off + bytes + 255) & ~size_t(255); return p; };

    unsigned short* w_cat_b  = (unsigned short*)alloc((size_t)GN * Hdim * 2);      // 12.6 MB
    unsigned short* fcn_w1_b = (unsigned short*)alloc((size_t)Hdim * DIN * 2);
    unsigned short* fcn_w2_b = (unsigned short*)alloc((size_t)Hdim * Hdim * 2);
    unsigned short* out_w_b  = (unsigned short*)alloc((size_t)DOUT * Hdim * 2);
    unsigned short* x_b      = (unsigned short*)alloc((size_t)Bsz * DIN * 2);
    float*          t1_f     = (float*)alloc((size_t)Bsz * Hdim * 4);
    unsigned short* t1_b     = (unsigned short*)alloc((size_t)Bsz * Hdim * 2);
    float*          h_f      = (float*)alloc((size_t)Bsz * Hdim * 4);
    unsigned short* h_b      = (unsigned short*)alloc((size_t)Bsz * Hdim * 2);
    float*          gates_f  = (float*)alloc((size_t)Bsz * GN * 4);                // 6.3 MB

    float* rec    = (float*)d_out;                                // [B, T, DOUT]
    float* latent = rec + (size_t)Bsz * Tlen * DOUT;              // [B, H]

    auto cvt = [&](const float* src, unsigned short* dst, int n) {
        k_f32_to_bf16<<<(n + 255) / 256, 256, 0, stream>>>(src, dst, n);
    };

    // --- one-time (per launch) bf16 casts ---
    cvt(x,      x_b,      Bsz * DIN);
    cvt(fcn_w1, fcn_w1_b, Hdim * DIN);
    cvt(fcn_w2, fcn_w2_b, Hdim * Hdim);
    cvt(w_ih,   w_cat_b,                       3 * Hdim * Hdim);
    cvt(w_hh,   w_cat_b + (size_t)3 * Hdim * Hdim, 3 * Hdim * Hdim);
    cvt(out_w,  out_w_b,  DOUT * Hdim);

    // --- FCN: latent = (x @ w1^T + b1) @ w2^T + b2 ---
    {
        dim3 g1(Hdim / 256, Bsz / 64);
        k_gemm_bf16_tn<<<g1, 256, 0, stream>>>((const __bf16*)x_b, (const __bf16*)fcn_w1_b,
                                               t1_f, fcn_b1, DIN, Hdim);
        cvt(t1_f, t1_b, Bsz * Hdim);
        dim3 g2(Hdim / 256, Bsz / 64);
        k_gemm_bf16_tn<<<g2, 256, 0, stream>>>((const __bf16*)t1_b, (const __bf16*)fcn_w2_b,
                                               h_f, fcn_b2, Hdim, Hdim);
        k_h_init<<<(Bsz * Hdim + 255) / 256, 256, 0, stream>>>(h_f, h_b, latent, Bsz * Hdim);
    }

    // --- GRU recurrence, out-projection fused per step ---
    dim3 gGate(GN / 256,  Bsz / 64);    // 24 x 4 blocks
    dim3 gOut (DOUT / 256, Bsz / 64);   //  2 x 4 blocks
    for (int t = 0; t < Tlen; ++t) {
        // gates = h @ [w_ih; w_hh]^T   (bias added in gating kernel)
        k_gemm_bf16_tn<<<gGate, 256, 0, stream>>>((const __bf16*)h_b, (const __bf16*)w_cat_b,
                                                  gates_f, nullptr, Hdim, GN);
        // h <- GRU(h, gates); also refresh bf16 copy
        k_gru_gate<<<(Bsz * Hdim) / 256, 256, 0, stream>>>(gates_f, b_ih, b_hh, h_f, h_b);
        // rec[:, t, :] = h @ out_w^T + out_b   (ldc = T*DOUT strides over time)
        k_gemm_bf16_tn<<<gOut, 256, 0, stream>>>((const __bf16*)h_b, (const __bf16*)out_w_b,
                                                 rec + (size_t)t * DOUT, out_b,
                                                 Hdim, Tlen * DOUT);
    }

    (void)in_sizes; (void)n_in; (void)out_size; (void)ws_size;
}
